// VectorQuantizer_4054449128245
// MI455X (gfx1250) — matrix operand order repaired and checked by reading, not compiled, not run on hardware
//
#include <hip/hip_runtime.h>

typedef __attribute__((ext_vector_type(8)))  _Float16 v8h;
typedef __attribute__((ext_vector_type(16))) _Float16 v16h;
typedef __attribute__((ext_vector_type(8)))  float    v8f;

#define KCB    8192       // codebook size
#define CDIM   256        // channels
#define NROWS  16384      // B*H*W
#define HWN    1024       // H*W
#define ROWS_PER_BLOCK 128
#define NTILE  32         // codebook entries per LDS buffer (x2 buffers)
#define LDSS   264        // padded LDS row stride (halves) -> conflict-free ds_load_b128

union V16 { v16h v; v8h h[2]; };

// ---------------- prep: codebook -> f16*(-2), e_sq in f32 ----------------
__global__ __launch_bounds__(256)
void vq_prep_codebook(const float* __restrict__ cb, _Float16* __restrict__ cbh,
                      float* __restrict__ esq) {
    __shared__ float red[256];
    const int k = blockIdx.x, c = threadIdx.x;
    float v = cb[(size_t)k * CDIM + c];
    cbh[(size_t)k * CDIM + c] = (_Float16)(-2.0f * v);
    red[c] = v * v;
    __syncthreads();
    for (int s = 128; s > 0; s >>= 1) {
        if (c < s) red[c] += red[c + s];
        __syncthreads();
    }
    if (c == 0) esq[k] = red[0];
}

// ---------------- prep: z (B,C,H,W) -> packed (N=B*HW, C) f16 ----------------
__global__ __launch_bounds__(256)
void vq_prep_z(const float* __restrict__ z, _Float16* __restrict__ zh) {
    const int bc = blockIdx.x;          // b*256 + c
    const int b = bc >> 8, c = bc & 255;
    const float* src = z + (size_t)b * CDIM * HWN + (size_t)c * HWN;
    for (int n = threadIdx.x; n < HWN; n += 256)
        zh[((size_t)(b * HWN + n)) * CDIM + c] = (_Float16)src[n];
}

// ---------------- main: fused GEMM(-2 z.eT) + argmin + gather + loss ----------------
__global__ __launch_bounds__(256)
void vq_main(const _Float16* __restrict__ zh, const _Float16* __restrict__ cbh,
             const float* __restrict__ esq, const float* __restrict__ cb,
             const float* __restrict__ z, float* __restrict__ out,
             float* __restrict__ loss_part) {
    __shared__ _Float16 s_cb[2][NTILE * LDSS];    // double-buffered codebook tile (33.8 KB)
    __shared__ float s_bd[8 * 16 * 16];
    __shared__ int   s_bi[8 * 16 * 16];
    __shared__ int   s_code[ROWS_PER_BLOCK];
    __shared__ float s_loss[256];

    const int tid   = threadIdx.x;
    const int wave  = tid >> 5;
    const int lane  = tid & 31;
    const int lhalf = lane >> 4;                  // 0 | 1 (selects K sub-range)
    const int l16   = lane & 15;
    const int blockRow0 = blockIdx.x * ROWS_PER_BLOCK;
    const int rbase = blockRow0 + wave * 16;

    // per-thread async-fill geometry: 4 chunks of b128 per buffer fill
    const int f_row = tid >> 3;                   // reuse below via chunk math instead
    (void)f_row;

    // ---- A matrix (16 z rows) resident in registers, WMMA 16x32 f16 layout ----
    v16h a[8];
    {
        const _Float16* zrow = zh + (size_t)(rbase + l16) * CDIM;
#pragma unroll
        for (int t = 0; t < 8; ++t) {
            const int off = t * 32 + lhalf * 8;
            V16 tmp;
            tmp.h[0] = *(const v8h*)(zrow + off);
            tmp.h[1] = *(const v8h*)(zrow + off + 16);
            a[t] = tmp.v;
        }
    }

    float best[8];
    int   bidx[8];
#pragma unroll
    for (int i = 0; i < 8; ++i) { best[i] = 3.4e38f; bidx[i] = 0; }

    // async fill of one codebook tile buffer: CDNA5 GLOBAL_LOAD_ASYNC_TO_LDS
    // (GVS mode: SGPR base + per-lane 32-bit byte offset; tracked by ASYNCcnt)
    auto fill_async = [&](int buf, int colb) {
#pragma unroll
        for (int p = 0; p < 4; ++p) {
            const int chunk = tid + p * 256;      // 0..1023 = 32 rows x 32 b128-chunks
            const int row = chunk >> 5;
            const int fb  = (chunk & 31) * 8;     // half index, 16B aligned
            const unsigned goff  = (unsigned)(((colb + row) * CDIM + fb) * 2);
            const unsigned laddr = (unsigned)(uintptr_t)&s_cb[buf][row * LDSS + fb];
            asm volatile("global_load_async_to_lds_b128 %0, %1, %2"
                         :: "v"(laddr), "v"(goff), "s"(cbh) : "memory");
        }
    };

    fill_async(0, 0);                              // prologue fill
    asm volatile("s_wait_asynccnt 0x0" ::: "memory");
    __syncthreads();

    for (int it = 0; it < KCB / NTILE; ++it) {
        const int colbase = it * NTILE;
        const int cur = it & 1;
        if (it + 1 < KCB / NTILE)                  // overlap next fill with WMMAs
            fill_async(cur ^ 1, colbase + NTILE);

#pragma unroll
        for (int sub = 0; sub < NTILE / 16; ++sub) {
            v8f acc = {};
            const _Float16* brow = &s_cb[cur][(sub * 16 + l16) * LDSS];
#pragma unroll
            for (int t = 0; t < 8; ++t) {
                const int off = t * 32 + lhalf * 8;
                V16 tmp;
                tmp.h[0] = *(const v8h*)(brow + off);
                tmp.h[1] = *(const v8h*)(brow + off + 16);
                acc = __builtin_amdgcn_wmma_f32_16x16x32_f16(
                          false, a[t], false, tmp.v, (short)0, acc, false, false);
            }
            const int coln = colbase + sub * 16 + l16;
            const float e2 = esq[coln];
#pragma unroll
            for (int i = 0; i < 8; ++i) {
                const float d = acc[i] + e2;       // = e^2 - 2 z.e
                if (d < best[i]) { best[i] = d; bidx[i] = coln; }
            }
        }

        asm volatile("s_wait_asynccnt 0x0" ::: "memory");
        __syncthreads();
    }

    // ---- cross-lane argmin via LDS (D layout: VGPR i -> row lhalf*8+i, col l16) ----
#pragma unroll
    for (int i = 0; i < 8; ++i) {
        const int r = wave * 16 + lhalf * 8 + i;
        s_bd[r * 16 + l16] = best[i];
        s_bi[r * 16 + l16] = bidx[i];
    }
    __syncthreads();

    if (lane < 16) {
        const int r = wave * 16 + lane;
        float bd = s_bd[r * 16];
        int   bi = s_bi[r * 16];
        for (int j = 1; j < 16; ++j) {
            const float d = s_bd[r * 16 + j];
            const int   x = s_bi[r * 16 + j];
            if (d < bd || (d == bd && x < bi)) { bd = d; bi = x; }
        }
        s_code[r] = bi;
        out[blockRow0 + r] = (float)bi;            // codes output
    }
    __syncthreads();

    // ---- gather quantized rows, write (B,C,H,W) output, loss partial ----
    float partial = 0.0f;
    float* qout = out + NROWS;
    for (int e = tid; e < ROWS_PER_BLOCK * CDIM; e += 256) {
        const int rloc = e & 127;
        const int c    = e >> 7;
        const int grow = blockRow0 + rloc;
        const float q  = cb[(size_t)s_code[rloc] * CDIM + c];
        const int b = grow >> 10, n = grow & 1023;
        const size_t zi = (size_t)b * CDIM * HWN + (size_t)c * HWN + n;
        const float df = q - z[zi];
        partial += df * df;
        qout[zi] = q;                              // straight-through value == q
    }
    s_loss[tid] = partial;
    __syncthreads();
    for (int s = 128; s > 0; s >>= 1) {
        if (tid < s) s_loss[tid] += s_loss[tid + s];
        __syncthreads();
    }
    if (tid == 0) loss_part[blockIdx.x] = s_loss[0];
}

// ---------------- deterministic loss finish ----------------
__global__ __launch_bounds__(128)
void vq_loss_final(const float* __restrict__ loss_part, float* __restrict__ out) {
    __shared__ float red[128];
    red[threadIdx.x] = loss_part[threadIdx.x];
    __syncthreads();
    for (int s = 64; s > 0; s >>= 1) {
        if (threadIdx.x < s) red[threadIdx.x] += red[threadIdx.x + s];
        __syncthreads();
    }
    if (threadIdx.x == 0)
        out[(size_t)NROWS + (size_t)NROWS * CDIM] =
            red[0] * (1.25f / (float)((size_t)NROWS * CDIM));
}

extern "C" void kernel_launch(void* const* d_in, const int* in_sizes, int n_in,
                              void* d_out, int out_size, void* d_ws, size_t ws_size,
                              hipStream_t stream) {
    const float* z  = (const float*)d_in[0];   // (16,256,32,32) f32
    const float* cb = (const float*)d_in[1];   // (8192,256) f32
    float* out = (float*)d_out;                // codes(16384) | quant(4194304) | loss(1)

    _Float16* ws_cbh = (_Float16*)d_ws;                                    // 4 MB
    _Float16* ws_zh  = (_Float16*)((char*)d_ws + (size_t)KCB * CDIM * 2);  // 8 MB
    float*    ws_esq = (float*)((char*)d_ws + (size_t)KCB * CDIM * 2
                                            + (size_t)NROWS * CDIM * 2);   // 32 KB
    float*    ws_lp  = ws_esq + KCB;                                       // 512 B

    vq_prep_codebook<<<KCB, 256, 0, stream>>>(cb, ws_cbh, ws_esq);
    vq_prep_z<<<(16 * CDIM), 256, 0, stream>>>(z, ws_zh);
    vq_main<<<NROWS / ROWS_PER_BLOCK, 256, 0, stream>>>(ws_zh, ws_cbh, ws_esq,
                                                        cb, z, out, ws_lp);
    vq_loss_final<<<1, 128, 0, stream>>>(ws_lp, out);
}
